// BitLlamaAttention_10376640987249
// MI455X (gfx1250) — matrix operand order repaired and checked by reading, not compiled, not run on hardware
//
#include <hip/hip_runtime.h>
#include <hip/hip_bf16.h>
#include <stdint.h>

typedef unsigned short u16;
typedef unsigned int   u32;
typedef __attribute__((ext_vector_type(16))) __bf16 v16bf;
typedef __attribute__((ext_vector_type(8)))  float  v8f;
typedef __attribute__((ext_vector_type(4)))  unsigned int v4u;
typedef __attribute__((ext_vector_type(8)))  int v8i;
typedef __attribute__((ext_vector_type(4)))  int v4i;

#define HIDDEN 2048
#define SEQ    2048
#define NHEADS 16
#define HDIM   128

#if defined(__HIP_DEVICE_COMPILE__) && __has_builtin(__builtin_amdgcn_tensor_load_to_lds)
#define HAVE_TDM 1
#else
#define HAVE_TDM 0
#endif

#if defined(__HIP_DEVICE_COMPILE__) && __has_builtin(__builtin_amdgcn_sched_barrier)
#define SCHED_FENCE() __builtin_amdgcn_sched_barrier(0)
#else
#define SCHED_FENCE()
#endif

__device__ inline u16 f2bf(float f) {
    u32 u = __float_as_uint(f);
    u32 r = u + 0x7FFFu + ((u >> 16) & 1u);   // round-to-nearest-even
    return (u16)(r >> 16);
}
__device__ inline float bf2f(u16 h) { return __uint_as_float(((u32)h) << 16); }

union FragBF { uint4 q[2]; v16bf v; };

// A-matrix 16x32 bf16 fragment (ISA 7.12.2): lane L: row = L%16;
// VGPR0-3 hold K = kb..kb+7, VGPR4-7 hold K = kb+16..kb+23, kb = (L>=16)*8
__device__ inline v16bf load_a_frag(const u16* base, int ld, int row0, int k0) {
    int lane = threadIdx.x & 31;
    int m  = lane & 15;
    int kb = (lane >> 4) << 3;
    const u16* p = base + (size_t)(row0 + m) * ld + k0 + kb;
    FragBF f;
    f.q[0] = *(const uint4*)(p);
    f.q[1] = *(const uint4*)(p + 16);
    return f.v;
}

// B-matrix 32x16 bf16 fragment from row-major R[n][k] (B = R^T):
// lane L: col n = L%16; K = kb..kb+7 and kb+16..kb+23, kb = (L>=16)*8
__device__ inline v16bf load_b_frag(const u16* base, int ld, int n0, int k0) {
    int lane = threadIdx.x & 31;
    int n  = lane & 15;
    int kb = (lane >> 4) << 3;
    const u16* p = base + (size_t)(n0 + n) * ld + k0 + kb;
    FragBF f;
    f.q[0] = *(const uint4*)(p);
    f.q[1] = *(const uint4*)(p + 16);
    return f.v;
}

__device__ inline v8f wmma_bf16(v16bf a, v16bf b, v8f c) {
    return __builtin_amdgcn_wmma_f32_16x16x32_bf16(false, a, false, b, (short)0, c, false, false);
}

// Generic LDS pointer: low 32 bits of the generic address are the LDS byte offset
__device__ inline u32 lds_byte_off(const void* p) { return (u32)(uintptr_t)p; }

#if HAVE_TDM
// TDM 2D tile load: 128 rows x 64 bf16 cols from a row-major tensor (row stride =
// 2048 elems), packed contiguously into LDS ([row][64], ld = 64 halves).
// D# packing per CDNA5 ISA 8.3/8.4.
__device__ inline void tdm_stage_128x64(u32 lds_off, const u16* tile_start) {
    unsigned long long ga = (unsigned long long)(uintptr_t)tile_start;
    v4u g0;
    g0[0] = 1u;                                            // count=1, user mode
    g0[1] = lds_off;                                       // lds_addr (bytes)
    g0[2] = (u32)(ga & 0xFFFFFFFFu);                       // global_addr[95:64]
    g0[3] = (u32)((ga >> 32) & 0x01FFFFFFu) | 0x80000000u; // addr[120:96] | type=2
    v8i g1;
    g1[0] = 0x00010000;                                    // data_size=1 (2B), no mcast
    g1[1] = (int)((2048u & 0xFFFFu) << 16);                // tensor_dim0 low16 @ [63:48]
    g1[2] = (int)((2048u & 0xFFFFu) << 16);                // tensor_dim1 low16 @ [95:80]
    g1[3] = (int)(64u << 16);                              // tile_dim0=64 @ [127:112]
    g1[4] = 128;                                           // tile_dim1=128, tile_dim2=0
    g1[5] = 2048;                                          // tensor_dim0_stride low32
    g1[6] = 0;
    g1[7] = 0;
    v4i z4 = {};
#if __clang_major__ >= 23
    v8i z8 = {};
    __builtin_amdgcn_tensor_load_to_lds(g0, g1, z4, z4, z8, 0);
#else
    __builtin_amdgcn_tensor_load_to_lds(g0, g1, z4, z4, 0);
#endif
}
#endif

// ---------- 1) groupwise ternary quantize fp32 -> bf16 (one wave per 128-group) ----------
__global__ void quant_kernel(const float* __restrict__ w, u16* __restrict__ out) {
    int gid  = (int)((blockIdx.x * blockDim.x + threadIdx.x) >> 5);
    int lane = threadIdx.x & 31;
    size_t base = (size_t)gid * 128 + (size_t)lane * 4;
    float4 x = *(const float4*)(w + base);
    float s = fabsf(x.x) + fabsf(x.y) + fabsf(x.z) + fabsf(x.w);
    #pragma unroll
    for (int off = 16; off > 0; off >>= 1) s += __shfl_xor(s, off, 32);
    float scale = fmaxf(s * (1.0f / 128.0f), 1e-8f);
    float th = 0.5f * scale;
    float v[4] = {x.x, x.y, x.z, x.w};
    u16 r[4];
    #pragma unroll
    for (int i = 0; i < 4; ++i) {
        float q = (v[i] > th) ? scale : ((v[i] < -th) ? -scale : 0.0f);
        r[i] = f2bf(q);
    }
    uint2 pk;
    pk.x = (u32)r[0] | ((u32)r[1] << 16);
    pk.y = (u32)r[2] | ((u32)r[3] << 16);
    *(uint2*)(out + base) = pk;
}

// ---------- 2) fp32 -> bf16 convert ----------
__global__ void tobf_kernel(const float* __restrict__ in, u16* __restrict__ out) {
    size_t i = ((size_t)blockIdx.x * blockDim.x + threadIdx.x) * 4;
    float4 v = *(const float4*)(in + i);
    uint2 pk;
    pk.x = (u32)f2bf(v.x) | ((u32)f2bf(v.y) << 16);
    pk.y = (u32)f2bf(v.z) | ((u32)f2bf(v.w) << 16);
    *(uint2*)(out + i) = pk;
}

// ---------- 3/6) GEMM: C[2048,2048] = A * W^T, TDM-staged LDS double buffering ----------
// Block: 256 thr (8 waves) -> 128x128 C tile; wave w owns rows w*16..w*16+15, all 128 N.
// K staged in 64-deep tiles: A(128x64) + B(128x64) bf16 in LDS, double buffered (64 KB).
// sched_barrier pins all 18 ds_loads before the 8 WMMAs -> countdown dscnt waits,
// WMMA overlaps remaining LDS traffic.
__global__ __launch_bounds__(256) void gemm_kernel(const u16* __restrict__ A,
                                                   const u16* __restrict__ W,
                                                   float* __restrict__ Cf,
                                                   u16* __restrict__ Ch, int f32out) {
    __shared__ u16 sA[2][128 * 64];
    __shared__ u16 sB[2][128 * 64];
    const int Kd = HIDDEN, N = HIDDEN;
    int wv   = threadIdx.x >> 5;
    int lane = threadIdx.x & 31;
    int mb = (int)(blockIdx.x >> 4) * 128;
    int nb = (int)(blockIdx.x & 15) * 128;
    v8f acc[8] = {};
    const int KT = Kd / 64;

#if HAVE_TDM
    if (wv == 0) {
        tdm_stage_128x64(lds_byte_off(&sA[0][0]), A + (size_t)mb * Kd);
        tdm_stage_128x64(lds_byte_off(&sB[0][0]), W + (size_t)nb * Kd);
    }
#else
    {
        int row = threadIdx.x >> 1, col = (threadIdx.x & 1) * 32;
        #pragma unroll
        for (int c = 0; c < 32; c += 8) {
            *(uint4*)&sA[0][row * 64 + col + c] = *(const uint4*)(A + (size_t)(mb + row) * Kd + col + c);
            *(uint4*)&sB[0][row * 64 + col + c] = *(const uint4*)(W + (size_t)(nb + row) * Kd + col + c);
        }
    }
#endif

    for (int kt = 0; kt < KT; ++kt) {
        int cur = kt & 1;
        if (kt + 1 < KT) {
            int k0 = (kt + 1) * 64;
#if HAVE_TDM
            if (wv == 0) {
                tdm_stage_128x64(lds_byte_off(&sA[cur ^ 1][0]), A + (size_t)mb * Kd + k0);
                tdm_stage_128x64(lds_byte_off(&sB[cur ^ 1][0]), W + (size_t)nb * Kd + k0);
            }
#else
            int row = threadIdx.x >> 1, col = (threadIdx.x & 1) * 32;
            #pragma unroll
            for (int c = 0; c < 32; c += 8) {
                *(uint4*)&sA[cur ^ 1][row * 64 + col + c] = *(const uint4*)(A + (size_t)(mb + row) * Kd + k0 + col + c);
                *(uint4*)&sB[cur ^ 1][row * 64 + col + c] = *(const uint4*)(W + (size_t)(nb + row) * Kd + k0 + col + c);
            }
#endif
        }
#if HAVE_TDM
        if (kt + 1 < KT) __builtin_amdgcn_s_wait_tensorcnt(2);
        else             __builtin_amdgcn_s_wait_tensorcnt(0);
#endif
        __syncthreads();
        #pragma unroll
        for (int ks = 0; ks < 64; ks += 32) {
            v16bf af = load_a_frag(&sA[cur][0], 64, wv * 16, ks);
            v16bf bf[8];
            #pragma unroll
            for (int j = 0; j < 8; ++j) bf[j] = load_b_frag(&sB[cur][0], 64, j * 16, ks);
            SCHED_FENCE();   // keep all fragment loads issued before the WMMA burst
            #pragma unroll
            for (int j = 0; j < 8; ++j) acc[j] = wmma_bf16(af, bf[j], acc[j]);
            SCHED_FENCE();
        }
        __syncthreads();   // all reads of `cur` done before it is overwritten
    }

    int hi = lane >> 4, cn = lane & 15;
    size_t rowb = (size_t)(mb + wv * 16 + 8 * hi) * N + nb + cn;
    if (f32out) {
        #pragma unroll
        for (int j = 0; j < 8; ++j)
            #pragma unroll
            for (int r = 0; r < 8; ++r)
                Cf[rowb + (size_t)r * N + j * 16] = acc[j][r];
    } else {
        #pragma unroll
        for (int j = 0; j < 8; ++j)
            #pragma unroll
            for (int r = 0; r < 8; ++r)
                Ch[rowb + (size_t)r * N + j * 16] = f2bf(acc[j][r]);
    }
}

// ---------- 4) RoPE + per-head relayout; V stored transposed [h][d][s] ----------
__global__ void rope_kernel(const u16* __restrict__ qraw, const u16* __restrict__ kraw,
                            const u16* __restrict__ vraw,
                            u16* __restrict__ Q, u16* __restrict__ K, u16* __restrict__ Vt) {
    int tid = blockIdx.x * blockDim.x + threadIdx.x;
    int j = tid & 63;
    int h = (tid >> 6) & 15;
    int s = tid >> 10;
    float freq = (float)s * __powf(10000.0f, -(float)j * (1.0f / 64.0f));
    float sn, c;
    __sincosf(freq, &sn, &c);
    size_t inb = (size_t)s * HIDDEN + (size_t)h * HDIM;
    float q1 = bf2f(qraw[inb + j]), q2 = bf2f(qraw[inb + j + 64]);
    float k1 = bf2f(kraw[inb + j]), k2 = bf2f(kraw[inb + j + 64]);
    size_t ob = (size_t)h * SEQ * HDIM + (size_t)s * HDIM;
    Q[ob + j]      = f2bf(q1 * c - q2 * sn);
    Q[ob + j + 64] = f2bf(q1 * sn + q2 * c);
    K[ob + j]      = f2bf(k1 * c - k2 * sn);
    K[ob + j + 64] = f2bf(k1 * sn + k2 * c);
    float v1 = bf2f(vraw[inb + j]), v2 = bf2f(vraw[inb + j + 64]);
    size_t vb = (size_t)h * HDIM * SEQ;
    Vt[vb + (size_t)j * SEQ + s]        = f2bf(v1);
    Vt[vb + (size_t)(j + 64) * SEQ + s] = f2bf(v2);
}

// ---------- 5) causal flash attention, one wave per 16-row query tile ----------
__global__ __launch_bounds__(256) void attn_kernel(const u16* __restrict__ Q,
                                                   const u16* __restrict__ K,
                                                   const u16* __restrict__ Vt,
                                                   const int* __restrict__ amask,
                                                   u16* __restrict__ Out) {
    __shared__ u16 Pbuf[8][16 * 32];
    int wv   = threadIdx.x >> 5;
    int lane = threadIdx.x & 31;
    int tile = blockIdx.x * 8 + wv;
    int head = tile >> 7;
    int m0   = (tile & 127) << 4;
    const u16* Qh = Q  + (size_t)head * SEQ * HDIM;
    const u16* Kh = K  + (size_t)head * SEQ * HDIM;
    const u16* Vh = Vt + (size_t)head * HDIM * SEQ;

    v16bf qf[4];
    #pragma unroll
    for (int i = 0; i < 4; ++i) qf[i] = load_a_frag(Qh, HDIM, m0, i * 32);

    v8f o[8] = {};
    float mrow[8], lrow[8];
    #pragma unroll
    for (int r = 0; r < 8; ++r) { mrow[r] = -1e30f; lrow[r] = 0.0f; }

    int hi = lane >> 4, cn = lane & 15;
    const float scale = 0.08838834764831845f;   // 1/sqrt(128)

    for (int t0 = 0; t0 <= m0 + 15; t0 += 32) {
        // ---- S = Q K^T : preload all 8 K-fragments, then 8 WMMAs back-to-back ----
        v16bf kb[8];
        #pragma unroll
        for (int kk = 0; kk < 4; ++kk) {
            kb[2 * kk]     = load_b_frag(Kh, HDIM, t0,      kk * 32);
            kb[2 * kk + 1] = load_b_frag(Kh, HDIM, t0 + 16, kk * 32);
        }
        SCHED_FENCE();
        v8f s0 = {}, s1 = {};
        #pragma unroll
        for (int kk = 0; kk < 4; ++kk) {
            s0 = wmma_bf16(qf[kk], kb[2 * kk],     s0);
            s1 = wmma_bf16(qf[kk], kb[2 * kk + 1], s1);
        }
        SCHED_FENCE();
        int tA = t0 + cn, tB = t0 + 16 + cn;
        int okA = amask[tA] != 0, okB = amask[tB] != 0;
        #pragma unroll
        for (int r = 0; r < 8; ++r) {
            int m = m0 + r + 8 * hi;
            float a = (tA <= m && okA) ? s0[r] * scale : -1e30f;
            float b = (tB <= m && okB) ? s1[r] * scale : -1e30f;
            float rmax = fmaxf(a, b);
            #pragma unroll
            for (int off = 1; off < 16; off <<= 1) rmax = fmaxf(rmax, __shfl_xor(rmax, off, 32));
            float mnew = fmaxf(mrow[r], rmax);
            float fac  = __expf(mrow[r] - mnew);
            float p0 = __expf(a - mnew), p1 = __expf(b - mnew);
            float rs = p0 + p1;
            #pragma unroll
            for (int off = 1; off < 16; off <<= 1) rs += __shfl_xor(rs, off, 32);
            lrow[r] = lrow[r] * fac + rs;
            mrow[r] = mnew;
            #pragma unroll
            for (int dt = 0; dt < 8; ++dt) o[dt][r] *= fac;
            int prow = r + 8 * hi;
            Pbuf[wv][prow * 32 + cn]      = f2bf(p0);
            Pbuf[wv][prow * 32 + 16 + cn] = f2bf(p1);
        }
        asm volatile("s_wait_dscnt 0x0" ::: "memory");  // LDS in-order per wave
        v16bf pa = load_a_frag(&Pbuf[wv][0], 32, 0, 0);
        // ---- O += P V : preload all 8 V-fragments, then 8 WMMAs back-to-back ----
        v16bf vb[8];
        #pragma unroll
        for (int dt = 0; dt < 8; ++dt) vb[dt] = load_b_frag(Vh, SEQ, dt * 16, t0);
        SCHED_FENCE();
        #pragma unroll
        for (int dt = 0; dt < 8; ++dt) o[dt] = wmma_bf16(pa, vb[dt], o[dt]);
        SCHED_FENCE();
    }
    #pragma unroll
    for (int r = 0; r < 8; ++r) {
        float inv = 1.0f / lrow[r];
        int srow = m0 + r + 8 * hi;
        #pragma unroll
        for (int dt = 0; dt < 8; ++dt)
            Out[(size_t)srow * HIDDEN + head * HDIM + dt * 16 + cn] = f2bf(o[dt][r] * inv);
    }
}

extern "C" void kernel_launch(void* const* d_in, const int* in_sizes, int n_in,
                              void* d_out, int out_size, void* d_ws, size_t ws_size,
                              hipStream_t stream) {
    const float* hidden = (const float*)d_in[0];
    const int*   amask  = (const int*)d_in[1];
    const float* wq     = (const float*)d_in[2];
    const float* wk     = (const float*)d_in[3];
    const float* wv     = (const float*)d_in[4];
    const float* wo     = (const float*)d_in[5];
    float* out = (float*)d_out;

    const size_t NE = (size_t)SEQ * HIDDEN;
    u16* ws   = (u16*)d_ws;
    u16* Xbf  = ws;
    u16* Wqb  = ws + NE * 1;
    u16* Wkb  = ws + NE * 2;
    u16* Wvb  = ws + NE * 3;
    u16* Wob  = ws + NE * 4;
    u16* qraw = ws + NE * 5;
    u16* kraw = ws + NE * 6;
    u16* vraw = ws + NE * 7;
    u16* Qr   = ws + NE * 8;
    u16* Kr   = ws + NE * 9;
    u16* Vt   = ws + NE * 10;
    u16* Attn = ws + NE * 11;                    // total 96 MiB scratch

    quant_kernel<<<4096, 256, 0, stream>>>(wq, Wqb);
    quant_kernel<<<4096, 256, 0, stream>>>(wk, Wkb);
    quant_kernel<<<4096, 256, 0, stream>>>(wv, Wvb);
    quant_kernel<<<4096, 256, 0, stream>>>(wo, Wob);
    tobf_kernel<<<4096, 256, 0, stream>>>(hidden, Xbf);

    gemm_kernel<<<256, 256, 0, stream>>>(Xbf, Wqb, nullptr, qraw, 0);
    gemm_kernel<<<256, 256, 0, stream>>>(Xbf, Wkb, nullptr, kraw, 0);
    gemm_kernel<<<256, 256, 0, stream>>>(Xbf, Wvb, nullptr, vraw, 0);

    rope_kernel<<<8192, 256, 0, stream>>>(qraw, kraw, vraw, Qr, Kr, Vt);
    attn_kernel<<<256, 256, 0, stream>>>(Qr, Kr, Vt, amask, Attn);

    gemm_kernel<<<256, 256, 0, stream>>>(Attn, Wob, out, nullptr, 1);
}
